// MultiHeadAttention_62045097558417
// MI455X (gfx1250) — matrix operand order repaired and checked by reading, not compiled, not run on hardware
//
#include <hip/hip_runtime.h>
#include <stdint.h>

// ---------------------------------------------------------------------------
// Multi-head attention for MI455X (gfx1250, wave32, WMMA f16 -> f32 accum)
//   x:[4,2048,768] fp32; W*:[768,768] fp32; out:[4,2048,768] fp32
// Pipeline: zero pads -> Q/K GEMMs (f16 [bh][t][64]) + V GEMM (f16 [bh][d][T],
//           pre-transposed) -> flash attention computing S^T/O^T so softmax
//           reductions are in-lane + one v_permlanex16 (no DS flush)
//           -> output projection GEMM.
// ---------------------------------------------------------------------------

typedef _Float16 v16h __attribute__((ext_vector_type(16)));
typedef _Float16 v8h  __attribute__((ext_vector_type(8)));
typedef _Float16 v4h  __attribute__((ext_vector_type(4)));
typedef float    v8f  __attribute__((ext_vector_type(8)));
typedef float    v4f  __attribute__((ext_vector_type(4)));

__device__ __forceinline__ v8f wmma16(v16h a, v16h b, v8f c) {
  // D = A(16x32 f16) * B(32x16 f16) + C(16x16 f32)
  return __builtin_amdgcn_wmma_f32_16x16x32_f16(false, a, false, b,
                                                (short)0, c, false, false);
}

// lane L <-> lane L^16 swap as a VALU op (identity selectors), avoiding
// ds_bpermute + s_wait_dscnt 0 pipeline flushes.
__device__ __forceinline__ float xor16f(float v) {
  const int i = __float_as_int(v);
  return __int_as_float(__builtin_amdgcn_permlanex16(
      i, i, 0x76543210, 0xfedcba98, false, false));
}

// A fragment (16x32 f16). ISA: lane<16 holds K 0-7 & 16-23 of row=lane,
// lane>=16 holds K 8-15 & 24-31 of row=lane-16. Two contiguous 16B loads.
__device__ __forceinline__ v16h frag_a_ld(const _Float16* p, int stride,
                                          int row0, int k0, int lane) {
  const int rs = lane & 15, hi = lane >> 4;
  const _Float16* r = p + (row0 + rs) * stride + k0;
  v8h lo = *(const v8h*)(r + hi * 8);
  v8h hf = *(const v8h*)(r + 16 + hi * 8);
  v16h o;
#pragma unroll
  for (int i = 0; i < 8; ++i) { o[i] = lo[i]; o[i + 8] = hf[i]; }
  return o;
}

// B fragment (32x16 f16) from K-contiguous storage p[col][k] (stride in halfs).
// lane = column (mod 16); lane<16 holds K 0-7 & 16-23, lane>=16 K 8-15 & 24-31.
__device__ __forceinline__ v16h frag_b_ld(const _Float16* p, int stride,
                                          int col0, int k0, int lane) {
  const int rs = lane & 15, hi = lane >> 4;
  const _Float16* c = p + (col0 + rs) * stride + k0 + hi * 8;
  v8h x = *(const v8h*)c;
  v8h y = *(const v8h*)(c + 16);
  v16h o;
#pragma unroll
  for (int i = 0; i < 8; ++i) { o[i] = x[i]; o[i + 8] = y[i]; }
  return o;
}

// ---------------------------------------------------------------------------
// GEMM: C[M=8192, N=768] = A[8192,768] @ W[768,768] + bias
//   A_F16: A is f16 (attention output) vs fp32 (x)
//   OUT_MODE 0: fp32 row-major (final output)
//   OUT_MODE 1: f16 head layout [bh][t][64] (d padded 48->64, pads pre-zeroed)
//   OUT_MODE 2: f16 transposed-V layout [bh][d(48)][2048]
// Block: 256 thr (8 waves), tile 128x128, BK=32; wave grid 2(M)x4(N).
// ---------------------------------------------------------------------------
template <bool A_F16, int OUT_MODE>
__global__ __launch_bounds__(256) void gemm_wmma(const void* __restrict__ Ap,
                                                 const float* __restrict__ W,
                                                 const float* __restrict__ bias,
                                                 void* __restrict__ Outp) {
  constexpr int BM = 128, BK = 32, LDA = BK + 8, LDB = BK + 8;
  __shared__ _Float16 As[BM * LDA];        // [m][k]
  __shared__ _Float16 Bs[128 * LDB];       // transposed: [n][k]
  const int tid = threadIdx.x;
  const int lane = tid & 31;
  const int w = tid >> 5;
  const int m0 = blockIdx.x * BM;
  const int n0 = blockIdx.y * 128;
  const int wm = (w >> 2) * 64;            // wave M offset: 0 or 64
  const int wn = (w & 3) * 32;             // wave N offset: 0..96

  v8f acc[4][2];
#pragma unroll
  for (int mi = 0; mi < 4; ++mi)
#pragma unroll
    for (int ni = 0; ni < 2; ++ni)
#pragma unroll
      for (int r = 0; r < 8; ++r) acc[mi][ni][r] = 0.0f;

  for (int k0 = 0; k0 < 768; k0 += BK) {
    __syncthreads();
    // ---- load A tile (128x32) into f16 LDS ----
    if (A_F16) {
      const _Float16* A = (const _Float16*)Ap;
      const int row = tid >> 1, hs = (tid & 1) << 4;
      const _Float16* src = A + (size_t)(m0 + row) * 768 + k0 + hs;
      *(v8h*)&As[row * LDA + hs]     = *(const v8h*)src;
      *(v8h*)&As[row * LDA + hs + 8] = *(const v8h*)(src + 8);
    } else {
      const float* A = (const float*)Ap;
#pragma unroll
      for (int i = 0; i < 4; ++i) {
        const int vid = i * 256 + tid;
        const int row = vid >> 3;
        const int c4 = (vid & 7) << 2;
        const v4f f = *(const v4f*)(A + (size_t)(m0 + row) * 768 + k0 + c4);
        v4h hv;
#pragma unroll
        for (int j = 0; j < 4; ++j) hv[j] = (_Float16)f[j];
        *(v4h*)&As[row * LDA + c4] = hv;        // single 8B ds_store
      }
    }
    // ---- load W tile (32x128 fp32) -> f16 LDS transposed to [n][k]:
    //      each thread moves a 4x4 sub-tile, transposing in registers ----
    {
      const int k4 = (tid >> 5) << 2;           // 0..28
      const int n4 = (tid & 31) << 2;           // 0..124
      v4f f[4];
#pragma unroll
      for (int j = 0; j < 4; ++j)
        f[j] = *(const v4f*)(W + (size_t)(k0 + k4 + j) * 768 + n0 + n4);
#pragma unroll
      for (int i = 0; i < 4; ++i) {
        v4h hv;
#pragma unroll
        for (int j = 0; j < 4; ++j) hv[j] = (_Float16)f[j][i];
        *(v4h*)&Bs[(n4 + i) * LDB + k4] = hv;   // single 8B ds_store
      }
    }
    __syncthreads();

    v16h av[4];
#pragma unroll
    for (int mi = 0; mi < 4; ++mi) av[mi] = frag_a_ld(As, LDA, wm + mi * 16, 0, lane);
    v16h bv[2];
#pragma unroll
    for (int ni = 0; ni < 2; ++ni) bv[ni] = frag_b_ld(Bs, LDB, wn + ni * 16, 0, lane);
#pragma unroll
    for (int mi = 0; mi < 4; ++mi)
#pragma unroll
      for (int ni = 0; ni < 2; ++ni)
        acc[mi][ni] = wmma16(av[mi], bv[ni], acc[mi][ni]);
  }

  // ---- epilogue: C layout = VGPR r -> row r+8*hi, lane&15 -> col ----
  const int rs = lane & 15, hi = lane >> 4;
#pragma unroll
  for (int mi = 0; mi < 4; ++mi)
#pragma unroll
    for (int ni = 0; ni < 2; ++ni)
#pragma unroll
      for (int r = 0; r < 8; ++r) {
        const int m = m0 + wm + mi * 16 + r + 8 * hi;
        const int n = n0 + wn + ni * 16 + rs;
        const float v = acc[mi][ni][r] + bias[n];
        if (OUT_MODE == 1) {                 // [bh][t][64] padded head layout
          const int b = m >> 11, t = m & 2047;
          const int h = n / 48, d = n - h * 48;
          ((_Float16*)Outp)[(((size_t)(b * 16 + h)) * 2048 + t) * 64 + d] =
              (_Float16)v;
        } else if (OUT_MODE == 2) {          // V^T: [bh][d(48)][2048]
          const int b = m >> 11, t = m & 2047;
          const int h = n / 48, d = n - h * 48;
          ((_Float16*)Outp)[(((size_t)(b * 16 + h)) * 48 + d) * 2048 + t] =
              (_Float16)v;
        } else {
          ((float*)Outp)[(size_t)m * 768 + n] = v;
        }
      }
}

// ---------------------------------------------------------------------------
// Flash attention, transposed orientation. Grid: (T/128, B*H), 256 thr.
// Wave w owns 16 query columns q = blockIdx.x*128 + w*16 + (lane&15).
// S^T = K @ Q^T  => each lane holds 32 key scores of ONE query in VGPRs:
// softmax max/sum are in-lane trees + one v_permlanex16. O^T = V^T @ P^T.
// ---------------------------------------------------------------------------
__global__ __launch_bounds__(256) void attn_flash(
    const _Float16* __restrict__ Qh,   // [bh][t][64]   (d 48..63 zero)
    const _Float16* __restrict__ Kh,   // [bh][t][64]   (d 48..63 zero)
    const _Float16* __restrict__ Vt,   // [bh][d(48)][2048]
    _Float16* __restrict__ Aatt) {     // [b*2048 + t][768] f16
  constexpr int LDK = 72;                    // padded LDS stride (16B-aligned)
  __shared__ _Float16 Ks[64 * LDK];          // [s][d]   -> A frags of K
  __shared__ _Float16 Vs[48 * LDK];          // [d][s]   -> A frags of V^T
  __shared__ _Float16 Pb[8 * 16 * LDK];      // per-wave P^T staging [q][s]

  const int tid = threadIdx.x, lane = tid & 31, w = tid >> 5;
  const int rs = lane & 15, hi = lane >> 4;
  const int bh = blockIdx.y;
  const int q0w = blockIdx.x * 128 + w * 16; // wave's first query
  const size_t qkbase = (size_t)bh * 2048 * 64;
  const size_t vbase  = (size_t)bh * 48 * 2048;

  // Q B-fragments (queries = columns), pre-scaled by 1/sqrt(48) in f16
  const _Float16 hscale = (_Float16)0.144337567f;
  v16h qb[2];
#pragma unroll
  for (int c = 0; c < 2; ++c) {
    qb[c] = frag_b_ld(Qh + qkbase, 64, q0w, c * 32, lane);
#pragma unroll
    for (int i = 0; i < 16; ++i) qb[c][i] *= hscale;
  }

  float m_run = -INFINITY, l_run = 0.0f;
  v8f ot[3];
#pragma unroll
  for (int di = 0; di < 3; ++di)
#pragma unroll
    for (int r = 0; r < 8; ++r) ot[di][r] = 0.0f;

  _Float16* Pl = &Pb[w * 16 * LDK];

  for (int s0 = 0; s0 < 2048; s0 += 64) {
    __syncthreads();                         // previous iter done reading LDS
    {   // K rows (64 x 64) row-major; V^T rows (48 x 64) straight copy
      const int row = tid >> 2, qs = (tid & 3) << 4;
      const _Float16* ksrc = Kh + qkbase + (size_t)(s0 + row) * 64 + qs;
      *(v8h*)&Ks[row * LDK + qs]     = *(const v8h*)ksrc;
      *(v8h*)&Ks[row * LDK + qs + 8] = *(const v8h*)(ksrc + 8);
      if (tid < 192) {
        const _Float16* vsrc = Vt + vbase + (size_t)row * 2048 + s0 + qs;
        *(v8h*)&Vs[row * LDK + qs]     = *(const v8h*)vsrc;
        *(v8h*)&Vs[row * LDK + qs + 8] = *(const v8h*)(vsrc + 8);
      }
    }
    __syncthreads();

    // S^T tiles: 4 key subtiles (M) x this wave's 16 queries (N)
    v8f st[4];
#pragma unroll
    for (int mi = 0; mi < 4; ++mi) {
#pragma unroll
      for (int r = 0; r < 8; ++r) st[mi][r] = 0.0f;
#pragma unroll
      for (int c = 0; c < 2; ++c)
        st[mi] = wmma16(frag_a_ld(Ks, LDK, mi * 16, c * 32, lane), qb[c], st[mi]);
    }

    // ---- online softmax: all 32 scores of query q=rs live in this lane ----
    float mt[4];
#pragma unroll
    for (int mi = 0; mi < 4; ++mi)           // 4 independent max trees
      mt[mi] = fmaxf(fmaxf(fmaxf(st[mi][0], st[mi][1]),
                           fmaxf(st[mi][2], st[mi][3])),
                     fmaxf(fmaxf(st[mi][4], st[mi][5]),
                           fmaxf(st[mi][6], st[mi][7])));
    float mloc = fmaxf(fmaxf(mt[0], mt[1]), fmaxf(mt[2], mt[3]));
    mloc = fmaxf(mloc, xor16f(mloc));        // combine lane halves (VALU)
    const float mnew = fmaxf(m_run, mloc);
    const float alpha = __expf(m_run - mnew);
    m_run = mnew;

    float sumt[4];
#pragma unroll
    for (int mi = 0; mi < 4; ++mi) {
#pragma unroll
      for (int r = 0; r < 8; ++r) st[mi][r] = __expf(st[mi][r] - mnew);
      sumt[mi] = ((st[mi][0] + st[mi][1]) + (st[mi][2] + st[mi][3])) +
                 ((st[mi][4] + st[mi][5]) + (st[mi][6] + st[mi][7]));
    }
    float ssum = (sumt[0] + sumt[1]) + (sumt[2] + sumt[3]);
    ssum += xor16f(ssum);
    l_run = l_run * alpha + ssum;
#pragma unroll
    for (int di = 0; di < 3; ++di)
#pragma unroll
      for (int r = 0; r < 8; ++r) ot[di][r] *= alpha;

    // stage P^T [q][s] : 4 packed 16B stores per lane (keys mi*16+8*hi+0..7)
#pragma unroll
    for (int mi = 0; mi < 4; ++mi) {
      v8h ph;
#pragma unroll
      for (int i = 0; i < 8; ++i) ph[i] = (_Float16)st[mi][i];
      *(v8h*)&Pl[rs * LDK + mi * 16 + 8 * hi] = ph;
    }

    // O^T += V^T @ P^T : d subtiles (M) x queries (N), contraction = 64 keys
#pragma unroll
    for (int c2 = 0; c2 < 2; ++c2) {
      const v16h pbf = frag_b_ld(Pl, LDK, 0, c2 * 32, lane);
#pragma unroll
      for (int di = 0; di < 3; ++di)
        ot[di] = wmma16(frag_a_ld(Vs, LDK, di * 16, c2 * 32, lane), pbf, ot[di]);
    }
  }

  // normalize (one l per lane) and store: 8 consecutive d per tile -> one 16B
  const int b = bh >> 4, h = bh & 15;
  const float rinv = 1.0f / l_run;
  const int q = q0w + rs;
  _Float16* orow = Aatt + (size_t)(b * 2048 + q) * 768 + h * 48;
#pragma unroll
  for (int di = 0; di < 3; ++di) {
    v8h hv;
#pragma unroll
    for (int i = 0; i < 8; ++i) hv[i] = (_Float16)(ot[di][i] * rinv);
    *(v8h*)(orow + di * 16 + 8 * hi) = hv;
  }
}

__global__ __launch_bounds__(256) void zero_f4(v4f* __restrict__ p, int n) {
  const int i = blockIdx.x * 256 + threadIdx.x;
  if (i < n) p[i] = (v4f){0.f, 0.f, 0.f, 0.f};
}

// ---------------------------------------------------------------------------
extern "C" void kernel_launch(void* const* d_in, const int* in_sizes, int n_in,
                              void* d_out, int out_size, void* d_ws,
                              size_t ws_size, hipStream_t stream) {
  (void)in_sizes; (void)n_in; (void)out_size; (void)ws_size;
  const float* x  = (const float*)d_in[0];
  const float* Wq = (const float*)d_in[1];
  const float* bq = (const float*)d_in[2];
  const float* Wk = (const float*)d_in[3];
  const float* bk = (const float*)d_in[4];
  const float* Wv = (const float*)d_in[5];
  const float* bv = (const float*)d_in[6];
  const float* Wp = (const float*)d_in[7];
  const float* bp = (const float*)d_in[8];

  // Workspace: Qh|Kh ([bh][2048][64] f16, 16 MB each, pads zeroed)
  //            Vt ([bh][48][2048] f16, 12 MB) | Aatt ([8192][768] f16, 12 MB)
  char* ws = (char*)d_ws;
  const size_t SZH = (size_t)64 * 2048 * 64 * sizeof(_Float16);   // 16 MB
  const size_t SZV = (size_t)64 * 48 * 2048 * sizeof(_Float16);   // 12 MB
  _Float16* Qh   = (_Float16*)(ws);
  _Float16* Kh   = (_Float16*)(ws + SZH);
  _Float16* Vt   = (_Float16*)(ws + 2 * SZH);
  _Float16* Aatt = (_Float16*)(ws + 2 * SZH + SZV);

  // zero Q/K head buffers (pad columns d=48..63 must be 0)
  const int nf4 = (int)(2 * SZH / 16);
  zero_f4<<<(nf4 + 255) / 256, 256, 0, stream>>>((v4f*)ws, nf4);

  const dim3 gg(64, 6), gb(256);
  gemm_wmma<false, 1><<<gg, gb, 0, stream>>>(x, Wq, bq, Qh);
  gemm_wmma<false, 1><<<gg, gb, 0, stream>>>(x, Wk, bk, Kh);
  gemm_wmma<false, 2><<<gg, gb, 0, stream>>>(x, Wv, bv, Vt);

  attn_flash<<<dim3(16, 64), gb, 0, stream>>>(Qh, Kh, Vt, Aatt);

  gemm_wmma<true, 0><<<gg, gb, 0, stream>>>(Aatt, Wp, bp, d_out);
}
